// Transformer_32796370272521
// MI455X (gfx1250) — compile-verified, matrix operand order checked by reading, not run on hardware
//
#include <hip/hip_runtime.h>

// ---------------- problem constants ----------------
constexpr int Bc = 2, Nc = 2048, Dc = 1024, Hc = 16, DHc = 64, DEPTHc = 8, Vc = 32000;
constexpr int INNERc = Hc * DHc;   // 1024
constexpr int FFc    = 4 * Dc;     // 4096
constexpr int Mc     = Bc * Nc;    // 4096 token rows

#define DEVI __device__ __forceinline__

typedef __attribute__((ext_vector_type(16))) _Float16 v16h;
typedef __attribute__((ext_vector_type(8)))  float    v8f;

// Load a 16x32 (A) or 32x16 (B) f16 WMMA fragment.  Caller passes a pointer
// already offset to (row_or_col * ld) + kbase + ((lane>>4)*8); halves j=0..7
// map to k..k+7 and j=8..15 map to k+16..k+23 per the CDNA5 lane layout.
DEVI v16h ldfrag(const _Float16* p) {
  v16h f;
  ((uint4*)&f)[0] = *(const uint4*)(p);
  ((uint4*)&f)[1] = *(const uint4*)(p + 16);
  return f;
}

DEVI v8f wmma16(v16h a, v16h b, v8f c) {
  return __builtin_amdgcn_wmma_f32_16x16x32_f16(false, a, false, b, (short)0, c,
                                                false, false);
}

// CDNA5 async global->LDS copy (ASYNCcnt-tracked, no VGPR data round trip).
// lds_byte = byte offset within the workgroup LDS allocation (flat addr low32).
DEVI void async_g2l_b128(unsigned lds_byte, const void* g) {
  asm volatile("global_load_async_to_lds_b128 %0, %1, off"
               :: "v"(lds_byte), "v"((unsigned long long)(uintptr_t)g)
               : "memory");
}
DEVI void wait_async0() {
#if __has_builtin(__builtin_amdgcn_s_wait_asynccnt)
  __builtin_amdgcn_s_wait_asynccnt(0);
#else
  asm volatile("s_wait_asynccnt 0" ::: "memory");
#endif
}

// reductions across the 16 lanes of each half-wave (C-layout column group)
DEVI float red16_max(float v) {
  v = fmaxf(v, __shfl_xor(v, 1, 32));
  v = fmaxf(v, __shfl_xor(v, 2, 32));
  v = fmaxf(v, __shfl_xor(v, 4, 32));
  v = fmaxf(v, __shfl_xor(v, 8, 32));
  return v;
}
DEVI float red16_sum(float v) {
  v += __shfl_xor(v, 1, 32);
  v += __shfl_xor(v, 2, 32);
  v += __shfl_xor(v, 4, 32);
  v += __shfl_xor(v, 8, 32);
  return v;
}

// ---------------- embedding ----------------
__global__ __launch_bounds__(256) void k_embed(const int* __restrict__ x,
                                               const float* __restrict__ tok,
                                               const float* __restrict__ pos,
                                               float* __restrict__ h) {
  int row = blockIdx.x;               // 0..Mc-1
  int n   = row & (Nc - 1);
  int id  = x[row];
  const float4* tp = (const float4*)(tok + (size_t)id * Dc);
  const float4* pp = (const float4*)(pos + (size_t)n * Dc);
  float4* hp = (float4*)(h + (size_t)row * Dc);
  int t = threadIdx.x;                // 256 threads * float4 = 1024 floats
  float4 a = tp[t], b = pp[t];
  hp[t] = make_float4(a.x + b.x, a.y + b.y, a.z + b.z, a.w + b.w);
}

// ------- weight convert + transpose: W[K][N] f32 -> Wt[N][K] f16, tiled -----
// grid: (N/32, K/32), 256 threads.
__global__ __launch_bounds__(256) void k_cvt_t(const float* __restrict__ W,
                                               _Float16* __restrict__ Wt,
                                               int K, int N) {
  __shared__ _Float16 tile[32][33];   // padded to dodge bank conflicts
  int n0 = blockIdx.x * 32, k0 = blockIdx.y * 32;
  int t = threadIdx.x;
  int kr = t >> 3;            // 0..31
  int nc = (t & 7) * 4;       // 0..28
  float4 w4 = *(const float4*)(W + (size_t)(k0 + kr) * N + n0 + nc);
  tile[kr][nc + 0] = (_Float16)w4.x;
  tile[kr][nc + 1] = (_Float16)w4.y;
  tile[kr][nc + 2] = (_Float16)w4.z;
  tile[kr][nc + 3] = (_Float16)w4.w;
  __syncthreads();
  int nr = t >> 3;            // 0..31
  int kc = (t & 7) * 4;       // 0..28
  _Float16 o4[4] = { tile[kc + 0][nr], tile[kc + 1][nr],
                     tile[kc + 2][nr], tile[kc + 3][nr] };
  *(uint2*)(Wt + (size_t)(n0 + nr) * K + k0 + kc) = *(uint2*)o4;
}

// ---------------- layernorm: fp32 in, f16 out (D=1024) ----------------
__global__ __launch_bounds__(256) void k_ln(const float* __restrict__ x,
                                            const float* __restrict__ w,
                                            const float* __restrict__ b,
                                            _Float16* __restrict__ out) {
  __shared__ float red[16];
  int row = blockIdx.x;
  const float4* xr = (const float4*)(x + (size_t)row * Dc);
  int t = threadIdx.x;
  float4 v4 = xr[t];
  float s1 = v4.x + v4.y + v4.z + v4.w;
  float s2 = v4.x * v4.x + v4.y * v4.y + v4.z * v4.z + v4.w * v4.w;
  for (int m = 1; m < 32; m <<= 1) {
    s1 += __shfl_xor(s1, m, 32);
    s2 += __shfl_xor(s2, m, 32);
  }
  int wave = t >> 5, lane = t & 31;
  if (lane == 0) { red[wave] = s1; red[8 + wave] = s2; }
  __syncthreads();
  if (t == 0) {
    float a = 0.f, c = 0.f;
    for (int i = 0; i < 8; ++i) { a += red[i]; c += red[8 + i]; }
    red[0] = a; red[8] = c;
  }
  __syncthreads();
  float mu = red[0] * (1.0f / Dc);
  float var = red[8] * (1.0f / Dc) - mu * mu;
  float rs = rsqrtf(var + 1e-5f);
  float4 wv = ((const float4*)w)[t];
  float4 bv = ((const float4*)b)[t];
  _Float16* op = out + (size_t)row * Dc + t * 4;
  op[0] = (_Float16)((v4.x - mu) * rs * wv.x + bv.x);
  op[1] = (_Float16)((v4.y - mu) * rs * wv.y + bv.y);
  op[2] = (_Float16)((v4.z - mu) * rs * wv.z + bv.z);
  op[3] = (_Float16)((v4.w - mu) * rs * wv.w + bv.w);
}

// ---------------- pipelined WMMA GEMM ----------------
// C[M x N] = A[M x K](f16) @ Bt[N x K](f16,transposed) (+bias) (+epilogue)
// mode 0: f16 out.  mode 1: f16 out = relu(acc+bias)^2.  mode 2: f32 out =
// acc + bias + resid (residual add, may alias out).
// Workgroup: 64 rows x 256 cols.  A tile double-buffered in LDS and filled
// with GLOBAL_LOAD_ASYNC_TO_LDS_B128 (overlaps WMMAs, no VGPR round trip);
// B fragments ping-pong in registers across a 2x-unrolled K loop (no moves).
// grid: (N/256, M/64), 256 threads.
__global__ __launch_bounds__(256) void k_gemm(const _Float16* __restrict__ A,
                                              const _Float16* __restrict__ Bt,
                                              const float* __restrict__ bias,
                                              const float* __restrict__ resid,
                                              void* __restrict__ Cout,
                                              int Nn, int Kk, int mode) {
  __shared__ __align__(16) _Float16 As[2][64 * 32];
  int tid = threadIdx.x;
  int wave = tid >> 5, lane = tid & 31;
  int mrow = lane & 15;
  int hi   = lane >> 4;
  int khi  = hi * 8;
  int m0 = blockIdx.y * 64;
  int n0 = blockIdx.x * 256 + wave * 32;

  int srow = tid >> 2;        // 0..63 stage row
  int sseg = (tid & 3) * 8;   // 0,8,16,24 halves
  const _Float16* astage = A + (size_t)(m0 + srow) * Kk + sseg;
  unsigned ldsA0 = (unsigned)(uintptr_t)&As[0][srow * 32 + sseg];
  unsigned ldsA1 = (unsigned)(uintptr_t)&As[1][srow * 32 + sseg];

  const _Float16* bp0 = Bt + (size_t)(n0 + mrow) * Kk + khi;
  const _Float16* bp1 = Bt + (size_t)(n0 + 16 + mrow) * Kk + khi;

  v8f acc[4][2] = {};

#define GEMM_COMPUTE(BUF, B0, B1)                                         \
  do {                                                                    \
    v16h a0_ = ldfrag(&As[BUF][(0 * 16 + mrow) * 32 + khi]);              \
    v16h a1_ = ldfrag(&As[BUF][(1 * 16 + mrow) * 32 + khi]);              \
    v16h a2_ = ldfrag(&As[BUF][(2 * 16 + mrow) * 32 + khi]);              \
    v16h a3_ = ldfrag(&As[BUF][(3 * 16 + mrow) * 32 + khi]);              \
    acc[0][0] = wmma16(a0_, B0, acc[0][0]);                               \
    acc[0][1] = wmma16(a0_, B1, acc[0][1]);                               \
    acc[1][0] = wmma16(a1_, B0, acc[1][0]);                               \
    acc[1][1] = wmma16(a1_, B1, acc[1][1]);                               \
    acc[2][0] = wmma16(a2_, B0, acc[2][0]);                               \
    acc[2][1] = wmma16(a2_, B1, acc[2][1]);                               \
    acc[3][0] = wmma16(a3_, B0, acc[3][0]);                               \
    acc[3][1] = wmma16(a3_, B1, acc[3][1]);                               \
  } while (0)

  // prologue: async-stage A(k=0) into buf0, prefetch B(k=0)
  async_g2l_b128(ldsA0, astage);
  v16h b0a = ldfrag(bp0);
  v16h b1a = ldfrag(bp1);
  wait_async0();
  __syncthreads();

  for (int k = 0; k < Kk; k += 64) {
    // half 1: stage A(k+32)->buf1, prefetch B(k+32); compute buf0 with b*a
    async_g2l_b128(ldsA1, astage + k + 32);
    v16h b0b = ldfrag(bp0 + k + 32);
    v16h b1b = ldfrag(bp1 + k + 32);
    __builtin_prefetch(bp0 + k + 256, 0, 1);
    __builtin_prefetch(bp1 + k + 256, 0, 1);
    GEMM_COMPUTE(0, b0a, b1a);
    wait_async0();
    __syncthreads();

    // half 2: stage A(k+64)->buf0, prefetch B(k+64); compute buf1 with b*b
    bool more = (k + 64) < Kk;
    if (more) {
      async_g2l_b128(ldsA0, astage + k + 64);
      b0a = ldfrag(bp0 + k + 64);
      b1a = ldfrag(bp1 + k + 64);
    }
    GEMM_COMPUTE(1, b0b, b1b);
    wait_async0();
    __syncthreads();
  }
#undef GEMM_COMPUTE

  _Float16* C16 = (_Float16*)Cout;
  float*    C32 = (float*)Cout;
#pragma unroll
  for (int t = 0; t < 2; ++t) {
    int col = n0 + t * 16 + mrow;
    float bb = bias ? bias[col] : 0.f;
#pragma unroll
    for (int u = 0; u < 4; ++u) {
#pragma unroll
      for (int r = 0; r < 8; ++r) {
        int row = m0 + u * 16 + r + 8 * hi;
        size_t idx = (size_t)row * Nn + col;
        float v = acc[u][t][r] + bb;
        if (mode == 0) {
          C16[idx] = (_Float16)v;
        } else if (mode == 1) {
          v = fmaxf(v, 0.f);
          C16[idx] = (_Float16)(v * v);
        } else {
          C32[idx] = v + resid[idx];
        }
      }
    }
  }
}

// ---------------- flash attention ----------------
// qkv: [Mc][3*INNER] f16 (q|k|v, heads interleaved by DH). o: [Mc][INNER] f16.
// grid: (B*H, N/64), 128 threads = 4 independent waves, each 16 query rows.
__global__ __launch_bounds__(128) void k_attn(const _Float16* __restrict__ qkv,
                                              _Float16* __restrict__ o) {
  constexpr int LDQ = 3 * INNERc;
  __shared__ __align__(16) _Float16 smem[4 * 2560];  // per wave: V^T 64x32 + P 16x32
  int bh = blockIdx.x;
  int b = bh / Hc, h = bh % Hc;
  int wave = threadIdx.x >> 5, lane = threadIdx.x & 31;
  int mrow = lane & 15;
  int hi   = lane >> 4;
  int khi  = hi * 8;
  int q0   = blockIdx.y * 64 + wave * 16;

  _Float16* vt = smem + wave * 2560;   // [dh][key] 64*32
  _Float16* pt = vt + 2048;            // [row][key] 16*32

  const _Float16* qp = qkv + (size_t)b * Nc * LDQ + h * DHc;
  const _Float16* kp = qp + INNERc;
  const _Float16* vp = qp + 2 * INNERc;

  v16h aq0 = ldfrag(qp + (size_t)(q0 + mrow) * LDQ + 0  + khi);
  v16h aq1 = ldfrag(qp + (size_t)(q0 + mrow) * LDQ + 32 + khi);

  float mi[8], li[8];
  v8f accO[4] = {};
#pragma unroll
  for (int r = 0; r < 8; ++r) { mi[r] = -1e30f; li[r] = 0.f; }

  const float scale = 0.125f;  // DH^-0.5
  int nblocks = (q0 + 15) / 32 + 1;
  for (int jb = 0; jb < nblocks; ++jb) {
    int kb = jb * 32;
    v8f s[2];
#pragma unroll
    for (int st = 0; st < 2; ++st) {
      int kcol = kb + st * 16 + mrow;
      v16h bk0 = ldfrag(kp + (size_t)kcol * LDQ + 0  + khi);
      v16h bk1 = ldfrag(kp + (size_t)kcol * LDQ + 32 + khi);
      v8f c = {};
      c = wmma16(aq0, bk0, c);
      c = wmma16(aq1, bk1, c);
      s[st] = c;
    }
    {
      const _Float16* vsrc = vp + (size_t)(kb + lane) * LDQ;
#pragma unroll 8
      for (int d = 0; d < 64; ++d) vt[d * 32 + lane] = vsrc[d];
    }
#pragma unroll
    for (int r = 0; r < 8; ++r) {
      int qrow = q0 + r + 8 * hi;
#pragma unroll
      for (int st = 0; st < 2; ++st) {
        int col = kb + st * 16 + mrow;
        float v = s[st][r] * scale;
        if (col > qrow) v = -3e38f;
        s[st][r] = v;
      }
      float tm = red16_max(fmaxf(s[0][r], s[1][r]));
      float mn = fmaxf(mi[r], tm);
      float alpha = __expf(mi[r] - mn);
      mi[r] = mn;
      float rs = 0.f;
#pragma unroll
      for (int st = 0; st < 2; ++st) {
        float e = __expf(s[st][r] - mn);
        s[st][r] = e;
        rs += e;
      }
      li[r] = li[r] * alpha + red16_sum(rs);
#pragma unroll
      for (int t = 0; t < 4; ++t) accO[t][r] *= alpha;
#pragma unroll
      for (int st = 0; st < 2; ++st)
        pt[(r + 8 * hi) * 32 + st * 16 + mrow] = (_Float16)s[st][r];
    }
    asm volatile("s_wait_dscnt 0" ::: "memory");
    v16h apf = ldfrag(pt + mrow * 32 + khi);
#pragma unroll
    for (int t = 0; t < 4; ++t) {
      v16h bv = ldfrag(vt + (t * 16 + mrow) * 32 + khi);
      accO[t] = wmma16(apf, bv, accO[t]);
    }
    asm volatile("s_wait_dscnt 0" ::: "memory");
  }
#pragma unroll
  for (int t = 0; t < 4; ++t) {
#pragma unroll
    for (int r = 0; r < 8; ++r) {
      int row = q0 + r + 8 * hi;
      int col = h * DHc + t * 16 + mrow;
      o[((size_t)b * Nc + row) * INNERc + col] = (_Float16)(accO[t][r] / li[r]);
    }
  }
}

// ---------------- fused logits + log-softmax + NLL ----------------
// ln: [Mc][D] f16.  Wt: [V][D] f16 (transposed logits_w).  bias: [V] f32.
// Per-lane online (m,l) accumulation; single cross-lane merge at the end.
// grid: Mc/16 blocks, 256 threads.
__global__ __launch_bounds__(256) void k_loss(const _Float16* __restrict__ ln,
                                              const _Float16* __restrict__ Wt,
                                              const float* __restrict__ bias,
                                              const int* __restrict__ labels,
                                              float* __restrict__ part) {
  __shared__ __align__(16) _Float16 As[16 * 1024];   // 32 KB
  __shared__ float sm_m[8][16], sm_l[8][16], sm_b[8][16];
  __shared__ float rown[16], rowm[16];
  int t0 = blockIdx.x * 16;
  {
    const uint4* src = (const uint4*)(ln + (size_t)t0 * Dc);
    uint4* dst = (uint4*)As;
    for (int i = threadIdx.x; i < 16 * 1024 / 8; i += 256) dst[i] = src[i];
  }
  __syncthreads();

  int wave = threadIdx.x >> 5, lane = threadIdx.x & 31;
  int mrow = lane & 15;
  int hi   = lane >> 4;
  int khi  = hi * 8;

  int labq[8];
  float mL[8], lL[8], labL[8];
#pragma unroll
  for (int r = 0; r < 8; ++r) {
    labq[r] = labels[t0 + r + 8 * hi];
    mL[r] = -3e38f; lL[r] = 0.f; labL[r] = -3e38f;
  }

  const _Float16* apb = As + mrow * 1024 + khi;
  for (int c0 = wave * 32; c0 < Vc; c0 += 256) {   // 125 sweeps, 2 tiles each
    const _Float16* bp0 = Wt + (size_t)(c0 + mrow) * Dc + khi;
    const _Float16* bp1 = Wt + (size_t)(c0 + 16 + mrow) * Dc + khi;
    v8f acc2[2] = {};
    for (int k = 0; k < Dc; k += 32) {
      v16h a = ldfrag(apb + k);
      acc2[0] = wmma16(a, ldfrag(bp0 + k), acc2[0]);
      acc2[1] = wmma16(a, ldfrag(bp1 + k), acc2[1]);
    }
#pragma unroll
    for (int tt = 0; tt < 2; ++tt) {
      int col = c0 + tt * 16 + mrow;
      float bv = bias[col];
#pragma unroll
      for (int r = 0; r < 8; ++r) {
        float sv = acc2[tt][r] + bv;
        if (col == labq[r]) labL[r] = sv;
        if (sv > mL[r]) {                   // per-lane online logsumexp
          lL[r] = lL[r] * __expf(mL[r] - sv) + 1.f;
          mL[r] = sv;
        } else {
          lL[r] += __expf(sv - mL[r]);
        }
      }
    }
  }
  // merge the 16 lanes of each half-wave once
#pragma unroll
  for (int r = 0; r < 8; ++r) {
    float m_ = mL[r], l_ = lL[r];
#pragma unroll
    for (int off = 1; off < 16; off <<= 1) {
      float mo = __shfl_xor(m_, off, 32);
      float lo = __shfl_xor(l_, off, 32);
      float mn = fmaxf(m_, mo);
      l_ = l_ * __expf(m_ - mn) + lo * __expf(mo - mn);
      m_ = mn;
    }
    mL[r] = m_; lL[r] = l_;
    labL[r] = red16_max(labL[r]);
  }

  if (mrow == 0) {
#pragma unroll
    for (int r = 0; r < 8; ++r) {
      int row = r + 8 * hi;
      sm_m[wave][row] = mL[r];
      sm_l[wave][row] = lL[r];
      sm_b[wave][row] = labL[r];
    }
  }
  __syncthreads();
  if (threadIdx.x < 16) {
    int row = threadIdx.x;
    float M = -3e38f;
    for (int w = 0; w < 8; ++w) M = fmaxf(M, sm_m[w][row]);
    float L = 0.f, Bst = -3e38f;
    for (int w = 0; w < 8; ++w) {
      L += sm_l[w][row] * __expf(sm_m[w][row] - M);
      Bst = fmaxf(Bst, sm_b[w][row]);
    }
    int label = labels[t0 + row];
    float msk = (label != 0) ? 1.f : 0.f;
    rown[row] = (M + __logf(L) - Bst) * msk;
    rowm[row] = msk;
  }
  __syncthreads();
  if (threadIdx.x == 0) {
    float sn = 0.f, sk = 0.f;
    for (int i = 0; i < 16; ++i) { sn += rown[i]; sk += rowm[i]; }
    part[blockIdx.x] = sn;
    part[256 + blockIdx.x] = sk;
  }
}

__global__ __launch_bounds__(256) void k_final(const float* __restrict__ part,
                                               float* __restrict__ out) {
  __shared__ float sa[8], sb[8];
  float a = part[threadIdx.x], b = part[256 + threadIdx.x];
  for (int mm = 1; mm < 32; mm <<= 1) {
    a += __shfl_xor(a, mm, 32);
    b += __shfl_xor(b, mm, 32);
  }
  int w = threadIdx.x >> 5, lane = threadIdx.x & 31;
  if (lane == 0) { sa[w] = a; sb[w] = b; }
  __syncthreads();
  if (threadIdx.x == 0) {
    float A = 0.f, Bm = 0.f;
    for (int i = 0; i < 8; ++i) { A += sa[i]; Bm += sb[i]; }
    out[0] = A / fmaxf(Bm, 1.f);
  }
}

// ---------------- host orchestration ----------------
extern "C" void kernel_launch(void* const* d_in, const int* in_sizes, int n_in,
                              void* d_out, int out_size, void* d_ws, size_t ws_size,
                              hipStream_t stream) {
  const int*   x      = (const int*)d_in[0];
  const int*   labels = (const int*)d_in[1];
  const float* tok    = (const float*)d_in[2];
  const float* pos    = (const float*)d_in[3];
  const float* anw    = (const float*)d_in[4];
  const float* anb    = (const float*)d_in[5];
  const float* qw     = (const float*)d_in[6];
  const float* aow    = (const float*)d_in[7];
  const float* aob    = (const float*)d_in[8];
  const float* fnw    = (const float*)d_in[9];
  const float* fnb    = (const float*)d_in[10];
  const float* fiw    = (const float*)d_in[11];
  const float* fib    = (const float*)d_in[12];
  const float* fow    = (const float*)d_in[13];
  const float* fob    = (const float*)d_in[14];
  const float* lnw    = (const float*)d_in[15];
  const float* lnb    = (const float*)d_in[16];
  const float* lw     = (const float*)d_in[17];
  const float* lb     = (const float*)d_in[18];
  (void)in_sizes; (void)n_in; (void)out_size; (void)ws_size;

  char* p = (char*)d_ws;
  auto carve = [&](size_t bytes) -> char* {
    char* r = p;
    p += (bytes + 255) & ~(size_t)255;
    return r;
  };
  float*    hbuf  = (float*)   carve((size_t)Mc * Dc * 4);       // residual stream
  _Float16* lnb16 = (_Float16*)carve((size_t)Mc * Dc * 2);       // LN output
  _Float16* qkvb  = (_Float16*)carve((size_t)Mc * 3 * INNERc * 2);
  _Float16* obuf  = (_Float16*)carve((size_t)Mc * INNERc * 2);
  _Float16* ffb   = (_Float16*)carve((size_t)Mc * FFc * 2);
  _Float16* wt    = (_Float16*)carve((size_t)Vc * Dc * 2);       // weight slab (max)
  float*    part  = (float*)   carve(512 * 4);

  k_embed<<<Mc, 256, 0, stream>>>(x, tok, pos, hbuf);

  for (int layer = 0; layer < DEPTHc; ++layer) {
    // --- attention block ---
    k_ln<<<Mc, 256, 0, stream>>>(hbuf, anw + layer * Dc, anb + layer * Dc, lnb16);
    size_t qn = (size_t)Dc * 3 * INNERc;
    k_cvt_t<<<dim3(3 * INNERc / 32, Dc / 32), 256, 0, stream>>>(
        qw + (size_t)layer * qn, wt, Dc, 3 * INNERc);
    k_gemm<<<dim3(3 * INNERc / 256, Mc / 64), 256, 0, stream>>>(
        lnb16, wt, nullptr, nullptr, (void*)qkvb, 3 * INNERc, Dc, 0);
    k_attn<<<dim3(Bc * Hc, Nc / 64), 128, 0, stream>>>(qkvb, obuf);
    size_t an = (size_t)INNERc * Dc;
    k_cvt_t<<<dim3(Dc / 32, INNERc / 32), 256, 0, stream>>>(
        aow + (size_t)layer * an, wt, INNERc, Dc);
    k_gemm<<<dim3(Dc / 256, Mc / 64), 256, 0, stream>>>(
        obuf, wt, aob + layer * Dc, hbuf, (void*)hbuf, Dc, INNERc, 2);
    // --- feedforward block ---
    k_ln<<<Mc, 256, 0, stream>>>(hbuf, fnw + layer * Dc, fnb + layer * Dc, lnb16);
    size_t fn = (size_t)Dc * FFc;
    k_cvt_t<<<dim3(FFc / 32, Dc / 32), 256, 0, stream>>>(
        fiw + (size_t)layer * fn, wt, Dc, FFc);
    k_gemm<<<dim3(FFc / 256, Mc / 64), 256, 0, stream>>>(
        lnb16, wt, fib + layer * FFc, nullptr, (void*)ffb, FFc, Dc, 1);
    k_cvt_t<<<dim3(Dc / 32, FFc / 32), 256, 0, stream>>>(
        fow + (size_t)layer * fn, wt, FFc, Dc);
    k_gemm<<<dim3(Dc / 256, Mc / 64), 256, 0, stream>>>(
        ffb, wt, fob + layer * Dc, hbuf, (void*)hbuf, Dc, FFc, 2);
  }

  // --- final LN + fused logits/log-softmax/NLL ---
  k_ln<<<Mc, 256, 0, stream>>>(hbuf, lnw, lnb, lnb16);
  k_cvt_t<<<dim3(Vc / 32, Dc / 32), 256, 0, stream>>>(lw, wt, Dc, Vc);
  k_loss<<<Mc / 16, 256, 0, stream>>>(lnb16, wt, lb, labels, part);
  k_final<<<1, 256, 0, stream>>>(part, (float*)d_out);
}
